// GatedFusion_2362232013195
// MI455X (gfx1250) — matrix operand order repaired and checked by reading, not compiled, not run on hardware
//
#include <hip/hip_runtime.h>
#include <hip/hip_bf16.h>
#include <math.h>

// ---------------------------------------------------------------------------
// GatedFusion for MI455X (gfx1250, wave32, WMMA).
//   B=32, S=2048, H=1024.
//   Roofline: ~768MB of stream-once seq traffic => ~33us @ 23.3TB/s.
//   Gate MLP uses v_wmma_f32_16x16x32_f16 (f32 accumulate), two independent
//   accumulator chains per wave. Streaming tensors use TH=NT so they don't
//   thrash the 192MB L2 (weights/alpha stay resident instead).
// ---------------------------------------------------------------------------

typedef __attribute__((ext_vector_type(16))) _Float16 v16h;
typedef __attribute__((ext_vector_type(8)))  float    v8f;
typedef __attribute__((ext_vector_type(4)))  float    v4f;

#define B_   32
#define S_   2048
#define H_   1024
#define H6_  (6 * H_)

// ---------------------------------------------------------------------------
// Kernel 1: pool LayerNorms + gap + 6H gate LayerNorm + mask denom.
// One block per batch row, 256 threads.
// ---------------------------------------------------------------------------
__global__ __launch_bounds__(256) void gate_prep_kernel(
    const float* __restrict__ guided_pool, const float* __restrict__ free_pool,
    const float* __restrict__ macro_pool,  const float* __restrict__ mezzo_pool,
    const float* __restrict__ micro_pool,  const int*   __restrict__ mask,
    const float* __restrict__ macro_g, const float* __restrict__ macro_b,
    const float* __restrict__ micro_g, const float* __restrict__ micro_b,
    const float* __restrict__ mezzo_g, const float* __restrict__ mezzo_b,
    const float* __restrict__ guided_g, const float* __restrict__ guided_b,
    const float* __restrict__ free_g,   const float* __restrict__ free_b,
    const float* __restrict__ gate_g,   const float* __restrict__ gate_b,
    float* __restrict__ gn_out, float* __restrict__ denom_out)
{
    const int b   = blockIdx.x;
    const int tid = threadIdx.x;

    __shared__ float gate[H6_];
    __shared__ float red0[256];
    __shared__ float red1[256];

    auto block_reduce2 = [&](float s, float s2, float& outs, float& outs2) {
        red0[tid] = s; red1[tid] = s2;
        __syncthreads();
        for (int off = 128; off > 0; off >>= 1) {
            if (tid < off) { red0[tid] += red0[tid + off]; red1[tid] += red1[tid + off]; }
            __syncthreads();
        }
        outs = red0[0]; outs2 = red1[0];
        __syncthreads();   // all readers done before red0/red1 reused
    };

    auto ln_seg = [&](const float* __restrict__ x, const float* __restrict__ g,
                      const float* __restrict__ bb, int seg) {
        const float* xb = x + b * H_;
        float v[4], s = 0.f, s2 = 0.f;
        #pragma unroll
        for (int r = 0; r < 4; r++) {
            float t = xb[tid + r * 256];
            v[r] = t; s += t; s2 += t * t;
        }
        float ts, ts2;
        block_reduce2(s, s2, ts, ts2);
        const float mu  = ts * (1.f / H_);
        const float var = ts2 * (1.f / H_) - mu * mu;
        const float rs  = rsqrtf(var + 1e-5f);
        #pragma unroll
        for (int r = 0; r < 4; r++) {
            int i = tid + r * 256;
            gate[seg * H_ + i] = (v[r] - mu) * rs * g[i] + bb[i];
        }
    };

    // concat order in reference: [macro, micro, mezzo, guided, free, gap]
    ln_seg(macro_pool,  macro_g,  macro_b,  0);
    ln_seg(micro_pool,  micro_g,  micro_b,  1);
    ln_seg(mezzo_pool,  mezzo_g,  mezzo_b,  2);
    ln_seg(guided_pool, guided_g, guided_b, 3);
    ln_seg(free_pool,   free_g,   free_b,   4);
    __syncthreads();
    #pragma unroll
    for (int r = 0; r < 4; r++) {
        int i = tid + r * 256;
        gate[5 * H_ + i] = fabsf(gate[3 * H_ + i] - gate[4 * H_ + i]);
    }
    __syncthreads();

    // LayerNorm over 6H
    float s = 0.f, s2 = 0.f;
    #pragma unroll
    for (int r = 0; r < 24; r++) {
        float t = gate[tid + r * 256];
        s += t; s2 += t * t;
    }
    float ts, ts2;
    block_reduce2(s, s2, ts, ts2);
    const float mu  = ts * (1.f / H6_);
    const float var = ts2 * (1.f / H6_) - mu * mu;
    const float rs  = rsqrtf(var + 1e-5f);
    #pragma unroll
    for (int r = 0; r < 24; r++) {
        int i = tid + r * 256;
        gn_out[(size_t)b * H6_ + i] = (gate[i] - mu) * rs * gate_g[i] + gate_b[i];
    }

    // denom[b] = max(sum(mask[b,:]), 1)
    int ms = 0;
    #pragma unroll
    for (int r = 0; r < 8; r++) ms += mask[b * S_ + tid + r * 256];
    float td, td2;
    block_reduce2((float)ms, 0.f, td, td2);
    if (tid == 0) denom_out[b] = fmaxf(td, 1.f);
}

// ---------------------------------------------------------------------------
// WMMA GEMM: out[M=32, N=1024] = act(A[32,K] @ W[K,1024] + bias).
// One wave (32 threads) per 16x16 output tile; K stepped by 64 with two
// independent v_wmma_f32_16x16x32_f16 accumulator chains. grid = (N/16, M/16).
// Fragment layouts per CDNA5 ISA 7.12.2 (wave32).
// ---------------------------------------------------------------------------
template <int KTOT, bool DO_GELU>
__global__ __launch_bounds__(32) void gemm_wmma_kernel(
    const float* __restrict__ A, const float* __restrict__ W,
    const float* __restrict__ bias, float* __restrict__ out)
{
    const int lane  = threadIdx.x;        // 0..31, full wave, EXEC all ones
    const int ntile = blockIdx.x;         // 0..63
    const int mtile = blockIdx.y;         // 0..1
    const int m     = lane & 15;
    const int half  = lane >> 4;
    const int ncol  = ntile * 16 + (lane & 15);

    const float* __restrict__ Arow = A + (size_t)(mtile * 16 + m) * KTOT;

    auto load_a = [&](int kb) {
        v16h af;
        // A 16x32 f16: lane half 0 holds K 0-7 & 16-23, half 1 holds 8-15 & 24-31
        #pragma unroll
        for (int j = 0; j < 16; j++) {
            int kloc = ((j >> 3) << 4) + (half << 3) + (j & 7);
            af[j] = (_Float16)Arow[kb + kloc];
        }
        return af;
    };
    auto load_b = [&](int kb) {
        v16h bf;
        // B 32x16 f16: lane half 0 holds K 0-7 & 16-23, half 1 holds 8-15 & 24-31 (col = lane&15)
        #pragma unroll
        for (int j = 0; j < 16; j++) {
            int kloc = ((j >> 3) << 4) + (half << 3) + (j & 7);
            bf[j] = (_Float16)W[(size_t)(kb + kloc) * H_ + ncol];
        }
        return bf;
    };

    v8f c0 = {}, c1 = {};
    for (int kb = 0; kb < KTOT; kb += 64) {
        v16h a0 = load_a(kb);
        v16h b0 = load_b(kb);
        c0 = __builtin_amdgcn_wmma_f32_16x16x32_f16(false, a0, false, b0,
                                                    (short)0, c0, false, false);
        v16h a1 = load_a(kb + 32);
        v16h b1 = load_b(kb + 32);
        c1 = __builtin_amdgcn_wmma_f32_16x16x32_f16(false, a1, false, b1,
                                                    (short)0, c1, false, false);
    }

    const float bv = bias[ncol];
    // D f32 16x16: VGPR r -> row r + 8*half, col = lane&15
    #pragma unroll
    for (int r = 0; r < 8; r++) {
        int row = mtile * 16 + r + half * 8;
        float x = c0[r] + c1[r] + bv;
        float y;
        if (DO_GELU) y = 0.5f * x * (1.f + erff(x * 0.70710678118f));
        else         y = 1.f / (1.f + expf(-x));   // sigmoid (GATE_TEMP = 1)
        out[(size_t)row * H_ + ncol] = y;
    }
}

// ---------------------------------------------------------------------------
// Kernel 4: the bandwidth kernel (sets the ~33us roofline).
// fused = (alpha*guided + (1-alpha)*free) * mask; accumulate masked col sums.
// grid = (B, S/128), 256 threads; thread owns 4 consecutive H columns.
// Stream-once tensors (512MB in, 256MB out) use non-temporal hints so they
// don't evict the resident weights/alpha from the 192MB L2.
// ---------------------------------------------------------------------------
__global__ __launch_bounds__(256) void fuse_kernel(
    const float* __restrict__ guided, const float* __restrict__ freeq,
    const float* __restrict__ alpha,  const int* __restrict__ mask,
    float* __restrict__ out_seq, float* __restrict__ pool_acc)
{
    const int b   = blockIdx.x;
    const int sc  = blockIdx.y;
    const int tid = threadIdx.x;
    const int h4  = tid * 4;
    const int s0  = sc * 128;

    __shared__ float mk[128];
    if (tid < 128) mk[tid] = (float)mask[b * S_ + s0 + tid];
    __syncthreads();

    const v4f al = *(const v4f*)(alpha + (size_t)b * H_ + h4);
    v4f acc = {0.f, 0.f, 0.f, 0.f};

    size_t base = ((size_t)b * S_ + s0) * H_ + h4;
    for (int i = 0; i < 128; i++) {
        const float w = mk[i];
        const v4f g = __builtin_nontemporal_load((const v4f*)(guided + base));
        const v4f f = __builtin_nontemporal_load((const v4f*)(freeq + base));
        v4f r = (al * g + (1.f - al) * f) * w;
        __builtin_nontemporal_store(r, (v4f*)(out_seq + base));
        acc += r;
        base += H_;
    }

    float* p = pool_acc + (size_t)b * H_ + h4;
    atomicAdd(p + 0, acc.x);
    atomicAdd(p + 1, acc.y);
    atomicAdd(p + 2, acc.z);
    atomicAdd(p + 3, acc.w);
}

__global__ __launch_bounds__(256) void pool_final_kernel(
    const float* __restrict__ acc, const float* __restrict__ denom,
    float* __restrict__ out_pool)
{
    const int b   = blockIdx.x;
    const int tid = threadIdx.x;
    const float inv = 1.f / denom[b];
    #pragma unroll
    for (int r = 0; r < 4; r++) {
        int i = tid + r * 256;
        out_pool[(size_t)b * H_ + i] = acc[(size_t)b * H_ + i] * inv;
    }
}

// ---------------------------------------------------------------------------
// Launch
// ---------------------------------------------------------------------------
extern "C" void kernel_launch(void* const* d_in, const int* in_sizes, int n_in,
                              void* d_out, int out_size, void* d_ws, size_t ws_size,
                              hipStream_t stream) {
    (void)in_sizes; (void)n_in; (void)out_size; (void)ws_size;

    const float* guided_seq  = (const float*)d_in[0];
    const float* guided_pool = (const float*)d_in[1];
    const float* free_seq    = (const float*)d_in[2];
    const float* free_pool   = (const float*)d_in[3];
    const float* macro_pool  = (const float*)d_in[4];
    const float* mezzo_pool  = (const float*)d_in[5];
    const float* micro_pool  = (const float*)d_in[6];
    const int*   mask_macro  = (const int*)d_in[7];
    const float* macro_g  = (const float*)d_in[8];
    const float* macro_b  = (const float*)d_in[9];
    const float* micro_g  = (const float*)d_in[10];
    const float* micro_b  = (const float*)d_in[11];
    const float* mezzo_g  = (const float*)d_in[12];
    const float* mezzo_b  = (const float*)d_in[13];
    const float* guided_g = (const float*)d_in[14];
    const float* guided_b = (const float*)d_in[15];
    const float* free_g   = (const float*)d_in[16];
    const float* free_b   = (const float*)d_in[17];
    const float* gate_g   = (const float*)d_in[18];
    const float* gate_b   = (const float*)d_in[19];
    const float* w1       = (const float*)d_in[20];
    const float* b1       = (const float*)d_in[21];
    const float* w2       = (const float*)d_in[22];
    const float* b2       = (const float*)d_in[23];

    // workspace layout (floats)
    float* ws = (float*)d_ws;
    float* gn    = ws;                          // 32*6144
    float* h     = gn + (size_t)B_ * H6_;       // 32*1024
    float* alpha = h  + (size_t)B_ * H_;        // 32*1024
    float* acc   = alpha + (size_t)B_ * H_;     // 32*1024
    float* denom = acc + (size_t)B_ * H_;       // 32

    float* out_seq  = (float*)d_out;
    float* out_pool = out_seq + (size_t)B_ * S_ * H_;

    // 1. LayerNorms + gate input prep + mask denom
    gate_prep_kernel<<<B_, 256, 0, stream>>>(
        guided_pool, free_pool, macro_pool, mezzo_pool, micro_pool, mask_macro,
        macro_g, macro_b, micro_g, micro_b, mezzo_g, mezzo_b,
        guided_g, guided_b, free_g, free_b, gate_g, gate_b,
        gn, denom);

    // 2. h = gelu(gn @ w1 + b1)   [32 x 6144] @ [6144 x 1024]
    gemm_wmma_kernel<H6_, true><<<dim3(H_ / 16, B_ / 16), 32, 0, stream>>>(gn, w1, b1, h);

    // 3. alpha = sigmoid(h @ w2 + b2)   [32 x 1024] @ [1024 x 1024]
    gemm_wmma_kernel<H_, false><<<dim3(H_ / 16, B_ / 16), 32, 0, stream>>>(h, w2, b2, alpha);

    // 4. zero pool accumulator (graph-capture-safe), then the bandwidth kernel
    hipMemsetAsync(acc, 0, (size_t)B_ * H_ * sizeof(float), stream);
    fuse_kernel<<<dim3(B_, S_ / 128), 256, 0, stream>>>(
        guided_seq, free_seq, alpha, mask_macro, out_seq, acc);

    // 5. pool = acc / denom
    pool_final_kernel<<<B_, 256, 0, stream>>>(acc, denom, out_pool);
}
